// EnhancedGenomicEncoder_48850958025033
// MI455X (gfx1250) — matrix operand order repaired and checked by reading, not compiled, not run on hardware
//
#include <hip/hip_runtime.h>
#include <hip/hip_bf16.h>

// ---------------------------------------------------------------------------
// B=32768, G=24, F=3, D=160, H=8, DH=20, HIDDEN=256.  Output: [B,256] f32.
// ---------------------------------------------------------------------------
#define BATCH   32768
#define NG      24
#define ND      160
#define NQKV    480
#define NH      8
#define NDH     20
#define FLATD   3840   // 24*160
#define N1      512    // HIDDEN*2
#define N2      256
#define N3      256

typedef __bf16 bf16x16 __attribute__((ext_vector_type(16)));
typedef float  f32x8   __attribute__((ext_vector_type(8)));

// ---------------- workspace layout (bytes, all 32B aligned) ----------------
static constexpr size_t H_OFF    = 0;                                     // [B][3840] bf16
static constexpr size_t W1T_OFF  = (size_t)BATCH * FLATD * 2;             // packed w1^T  bf16
static constexpr size_t W2T_OFF  = W1T_OFF + (size_t)FLATD * N1 * 2;      // packed w2^T  bf16
static constexpr size_t W3T_OFF  = W2T_OFF + (size_t)N1 * N2 * 2;         // packed w3^T  bf16
static constexpr size_t OWT_OFF  = W3T_OFF + (size_t)N2 * N3 * 2;         // packed out_w^T bf16
static constexpr size_t QKVB_OFF = OWT_OFF + (size_t)ND * ND * 2;         // [24][480]   bf16
static constexpr size_t U_OFF    = QKVB_OFF + (size_t)NG * NQKV * 2;      // [2][480]    f32
static constexpr size_t AG_OFF   = U_OFF + 2 * NQKV * 4;                  // [24][160]   f32
static constexpr size_t E_OFF    = AG_OFF + (size_t)NG * ND * 4;          // [2][64]     f32

// ---------------- WMMA fragment helpers (bf16, 16x16x32) ----------------
// A: 16 rows x 32 K, row-major source with row stride ld.
// lane&15 = row; lane>>4 picks K-halves {0..15} vs {16..31}.
// Each half is 16 contiguous bytes -> compiler merges to ds/global b128.
__device__ __forceinline__ bf16x16 load_a_frag(const __bf16* base, int ld, int lane) {
    int r  = lane & 15;
    int hi = (lane >> 4) << 4;          // 0 or 16
    const __bf16* row = base + r * ld;
    bf16x16 a;
#pragma unroll
    for (int j = 0; j < 4; ++j) {
        int k0 = 2 * j + hi;
        a[2*j]       = row[k0];
        a[2*j + 1]   = row[k0 + 1];
        a[8 + 2*j]     = row[8 + k0];
        a[8 + 2*j + 1] = row[8 + k0 + 1];
    }
    return a;
}

// Packed-B: each 32x16 tile stored as the exact per-lane register image:
// 512 bf16 per tile, lane l owns 32 contiguous bytes.
// Element (k,n) of a tile lives at lane = ((k>>4)&1)*16 | (n&15), byte-slot k&15.
__device__ __forceinline__ size_t pk_idx(int k, int n, int ncols) {
    size_t tile = (size_t)((k >> 5) * (ncols >> 4) + (n >> 4));
    int lane = (((k >> 4) & 1) << 4) | (n & 15);
    return (tile << 9) + ((size_t)lane << 4) + (size_t)(k & 15);
}

__device__ __forceinline__ bf16x16 load_b_packed(const __bf16* base, int tileIdx, int lane) {
    const bf16x16* p = (const bf16x16*)(base + ((size_t)tileIdx << 9));
    return p[lane];
}

__device__ __forceinline__ f32x8 wmma_bf16(bf16x16 a, bf16x16 b, f32x8 c) {
    return __builtin_amdgcn_wmma_f32_16x16x32_bf16(false, a, false, b,
                                                   (short)0, c, false, false);
}

// ---------------------------------------------------------------------------
// Kernel P: pack weights to fragment-major bf16 + rank-2 attention tables.
// qkv[b,g] = qkv_base[g] + x0*u_bin + (x1+x2)*u_feat   (exact refactor)
// ---------------------------------------------------------------------------
__global__ __launch_bounds__(256) void prep_kernel(
    const float* __restrict__ gene_emb, const float* __restrict__ type_emb,
    const float* __restrict__ w_bin, const float* __restrict__ b_bin,
    const float* __restrict__ w_feat, const float* __restrict__ b_feat,
    const float* __restrict__ ipw, const float* __restrict__ ipb,
    const float* __restrict__ out_w,
    const float* __restrict__ w1, const float* __restrict__ w2,
    const float* __restrict__ w3,
    __bf16* __restrict__ w1p, __bf16* __restrict__ w2p,
    __bf16* __restrict__ w3p, __bf16* __restrict__ owp,
    __bf16* __restrict__ qkvb, float* __restrict__ u,
    float* __restrict__ ag, float* __restrict__ e)
{
    int tid = blockIdx.x * blockDim.x + threadIdx.x;
    int nth = gridDim.x * blockDim.x;

    for (int i = tid; i < FLATD * N1; i += nth) {
        int k = i >> 9, n = i & 511;
        w1p[pk_idx(k, n, N1)] = (__bf16)w1[n * FLATD + k];
    }
    for (int i = tid; i < N1 * N2; i += nth) {
        int k = i >> 8, n = i & 255;
        w2p[pk_idx(k, n, N2)] = (__bf16)w2[n * N1 + k];
    }
    for (int i = tid; i < N2 * N3; i += nth) {
        int k = i >> 8, n = i & 255;
        w3p[pk_idx(k, n, N3)] = (__bf16)w3[n * N2 + k];
    }
    for (int i = tid; i < ND * ND; i += nth) {
        int k = i / ND, n = i % ND;
        owp[pk_idx(k, n, ND)] = (__bf16)out_w[n * ND + k];
    }

    if (blockIdx.x == 0) {
        __shared__ float s_ag[NG * ND];
        __shared__ float s_e[128];
        int t = threadIdx.x;
        for (int i = t; i < 128; i += 256) {
            float v = (i < 64 ? w_bin[i] : w_feat[i - 64]) * (1.f / 3.f);
            s_e[i] = v; e[i] = v;
        }
        for (int i = t; i < NG * ND; i += 256) {
            int g = i / ND, d = i % ND;
            float v;
            if (d < 64) v = gene_emb[g * 64 + d];
            else if (d < 96) {
                int dd = d - 64;
                v = (type_emb[dd] + type_emb[32 + dd] + type_emb[64 + dd]) * (1.f / 3.f);
            } else {
                int dd = d - 96;
                v = (b_bin[dd] + 2.f * b_feat[dd]) * (1.f / 3.f);
            }
            s_ag[i] = v; ag[i] = v;
        }
        __syncthreads();
        for (int i = t; i < NG * NQKV; i += 256) {
            int g = i / NQKV, j = i % NQKV;
            const float* wr = ipw + j * ND;
            float acc = ipb[j];
            for (int d = 0; d < ND; ++d) acc += wr[d] * s_ag[g * ND + d];
            qkvb[i] = (__bf16)acc;
        }
        for (int i = t; i < 2 * NQKV; i += 256) {
            int which = i / NQKV, j = i % NQKV;
            const float* wr = ipw + j * ND + 96;
            const float* ev = s_e + which * 64;
            float acc = 0.f;
            for (int d = 0; d < 64; ++d) acc += wr[d] * ev[d];
            u[i] = acc;
        }
    }
}

// ---------------------------------------------------------------------------
// Kernel A: rank-2 QKV + WMMA attention (scores & ctx) + WMMA out-projection
//           + residual + LayerNorm.  16 batch / WG, 8 waves (2 batch / wave).
// Per-wave 8KB staging: [0,2K)=q/P rows (A-side, 32x32), [2K,4K)=k/v packed-B
// (2 tiles), [4K,8K)=f32 scores.  LDS total ~231 KB.
// ---------------------------------------------------------------------------
__global__ __launch_bounds__(256) void attn_kernel(
    const float* __restrict__ gf,
    const float* __restrict__ out_b,
    const float* __restrict__ ln_g, const float* __restrict__ ln_b,
    const __bf16* __restrict__ qkvb_g, const float* __restrict__ u_g,
    const float* __restrict__ ag_g, const float* __restrict__ e_g,
    const __bf16* __restrict__ owp, __bf16* __restrict__ h_out)
{
    __shared__ __bf16 s_qkv[NG * NQKV];          // 23040 B
    __shared__ float  s_u[2 * NQKV];             //  3840 B
    __shared__ float  s_ag[NG * ND];             // 15360 B
    __shared__ float  s_e[128];                  //   512 B
    __shared__ float  s_ln[2 * ND];              //  1280 B
    __shared__ float  s_ob[ND];                  //   640 B
    __shared__ float  s_s0[384];                 //  1536 B
    __shared__ float  s_s12[384];                //  1536 B
    __shared__ __attribute__((aligned(32))) unsigned char s_stage[8][8192]; // 65536 B
    __shared__ __bf16 s_ctx[384 * ND];           // 122880 B (ctx, then re-used as h)

    const int tid  = threadIdx.x;
    const int lane = tid & 31;
    const int w    = tid >> 5;
    const int b0   = blockIdx.x * 16;
    const f32x8 zero8 = {0.f, 0.f, 0.f, 0.f, 0.f, 0.f, 0.f, 0.f};

    // ---- cooperative loads + one-time zero of staging ----
    {
        const unsigned* src = (const unsigned*)qkvb_g;
        unsigned* dst = (unsigned*)s_qkv;
        for (int i = tid; i < NG * NQKV / 2; i += 256) dst[i] = src[i];
    }
    for (int i = tid; i < 2 * NQKV; i += 256) s_u[i] = u_g[i];
    for (int i = tid; i < NG * ND; i += 256) s_ag[i] = ag_g[i];
    for (int i = tid; i < 128; i += 256) s_e[i] = e_g[i];
    for (int i = tid; i < ND; i += 256) {
        s_ln[i] = ln_g[i]; s_ln[ND + i] = ln_b[i]; s_ob[i] = out_b[i];
    }
    {
        unsigned* z = (unsigned*)s_stage;
        for (int i = tid; i < 8 * 8192 / 4; i += 256) z[i] = 0u;
    }
    for (int r = tid; r < 384; r += 256) {
        int b = b0 + r / NG, g = r % NG;
        const float* p = gf + (size_t)b * (NG * 3) + g * 3;
        s_s0[r]  = p[0];
        s_s12[r] = p[1] + p[2];
    }
    __syncthreads();

    // ---- attention: wave w handles batch elements 2w, 2w+1 ----
    __bf16* ab = (__bf16*)(s_stage[w]);            // 32x32 row-major A-side (q / P)
    __bf16* bb = (__bf16*)(s_stage[w] + 2048);     // 2 packed B tiles (k / v)
    float*  sb = (float*)(s_stage[w] + 4096);      // 32x32 f32 scores

    for (int ib = 0; ib < 2; ++ib) {
        int bl = w * 2 + ib;
        for (int hh = 0; hh < NH; ++hh) {
            // -- build q (A rows) and k (packed B); restore zero pads --
            if (lane < NG) {
                int g = lane;
                int r = bl * NG + g;
                float a0 = s_s0[r], a12 = s_s12[r];
                const __bf16* base = s_qkv + g * NQKV;
#pragma unroll
                for (int d2 = 20; d2 < 24; ++d2) ab[g * 32 + d2] = (__bf16)0.f;   // stale P
                {
                    int nt = g >> 4, lt = 16 + (g & 15);
#pragma unroll
                    for (int kz = 20; kz < 24; ++kz)                              // stale v
                        bb[nt * 512 + lt * 16 + (kz & 15)] = (__bf16)0.f;
                }
#pragma unroll
                for (int d = 0; d < NDH; ++d) {
                    int j = hh * NDH + d;
                    float qv = (float)base[j]      + a0 * s_u[j]        + a12 * s_u[NQKV + j];
                    float kv = (float)base[ND + j] + a0 * s_u[ND + j]   + a12 * s_u[NQKV + ND + j];
                    ab[g * 32 + d] = (__bf16)qv;
                    int nt = g >> 4;                                   // element (k=d, n=g)
                    int lt = (((d >> 4) & 1) << 4) | (g & 15);
                    bb[nt * 512 + lt * 16 + (d & 15)] = (__bf16)kv;
                }
            }
            __builtin_amdgcn_wave_barrier();
            // -- scores = Q K^T / sqrt(20): 2x2 WMMA tiles --
#pragma unroll
            for (int mt = 0; mt < 2; ++mt) {
                bf16x16 a = load_a_frag(ab + mt * 16 * 32, 32, lane);
#pragma unroll
                for (int nt = 0; nt < 2; ++nt) {
                    bf16x16 b = load_b_packed(bb, nt, lane);
                    f32x8 acc = wmma_bf16(a, b, zero8);
                    int n = lane & 15, hi = lane >> 4;
#pragma unroll
                    for (int j = 0; j < 8; ++j) {
                        int m = mt * 16 + j + hi * 8;
                        sb[m * 32 + nt * 16 + n] = acc[j] * 0.22360679774997896f;
                    }
                }
            }
            __builtin_amdgcn_wave_barrier();
            // -- softmax -> P (A rows), build v (packed B) --
            if (lane < NG) {
                int g = lane;
                float sc[NG], m = -1e30f;
#pragma unroll
                for (int kk = 0; kk < NG; ++kk) { sc[kk] = sb[g * 32 + kk]; m = fmaxf(m, sc[kk]); }
                float sum = 0.f;
#pragma unroll
                for (int kk = 0; kk < NG; ++kk) { sc[kk] = __expf(sc[kk] - m); sum += sc[kk]; }
                float inv = 1.f / sum;
#pragma unroll
                for (int kk = 0; kk < NG; ++kk) ab[g * 32 + kk] = (__bf16)(sc[kk] * inv);
                int r = bl * NG + g;
                float a0 = s_s0[r], a12 = s_s12[r];
                const __bf16* base = s_qkv + g * NQKV;
#pragma unroll
                for (int d = 0; d < NDH; ++d) {
                    int j = hh * NDH + d;
                    float vv = (float)base[2*ND + j] + a0 * s_u[2*ND + j] + a12 * s_u[NQKV + 2*ND + j];
                    int nt = d >> 4;                                   // element (k=g, n=d)
                    int lt = (((g >> 4) & 1) << 4) | (d & 15);
                    bb[nt * 512 + lt * 16 + (g & 15)] = (__bf16)vv;
                }
            }
            __builtin_amdgcn_wave_barrier();
            // -- ctx = P V: 2x2 WMMA tiles --
#pragma unroll
            for (int mt = 0; mt < 2; ++mt) {
                bf16x16 a = load_a_frag(ab + mt * 16 * 32, 32, lane);
#pragma unroll
                for (int nt = 0; nt < 2; ++nt) {
                    bf16x16 b = load_b_packed(bb, nt, lane);
                    f32x8 acc = wmma_bf16(a, b, zero8);
                    int n = lane & 15, hi = lane >> 4;
#pragma unroll
                    for (int j = 0; j < 8; ++j) {
                        int m = mt * 16 + j + hi * 8;
                        int d = nt * 16 + n;
                        if (m < NG && d < NDH)
                            s_ctx[(bl * NG + m) * ND + hh * NDH + d] = (__bf16)acc[j];
                    }
                }
            }
            __builtin_amdgcn_wave_barrier();
        }
    }
    __syncthreads();

    // ---- out-projection: ctx(384x160) @ out_w^T, WMMA, overwrite ctx with h ----
    // wave w owns row-tiles {w, w+8, w+16}; all 10 col-tiles register-accumulated.
    for (int rr = 0; rr < 3; ++rr) {
        int rt = w + rr * 8;
        f32x8 acc[10];
#pragma unroll
        for (int i = 0; i < 10; ++i) acc[i] = zero8;
#pragma unroll
        for (int kk = 0; kk < 5; ++kk) {
            bf16x16 a = load_a_frag(s_ctx + (rt * 16) * ND + kk * 32, ND, lane);
#pragma unroll
            for (int ct = 0; ct < 10; ++ct) {
                bf16x16 b = load_b_packed(owp, kk * 10 + ct, lane);
                acc[ct] = wmma_bf16(a, b, acc[ct]);
            }
        }
        __builtin_amdgcn_wave_barrier();
        int n = lane & 15, hi = lane >> 4;
#pragma unroll
        for (int ct = 0; ct < 10; ++ct) {
#pragma unroll
            for (int j = 0; j < 8; ++j) {
                int r = rt * 16 + j + hi * 8;
                int d = ct * 16 + n;
                int g = r % NG;
                float res = s_ag[g * ND + d];
                if (d >= 96)
                    res += s_s0[r] * s_e[d - 96] + s_s12[r] * s_e[64 + d - 96];
                s_ctx[r * ND + d] = (__bf16)(acc[ct][j] + s_ob[d] + res);   // now h
            }
        }
        __builtin_amdgcn_wave_barrier();
    }
    __syncthreads();

    // ---- LayerNorm + store h (bf16), flat layout g*160+d ----
    for (int r = tid; r < 384; r += 256) {
        const __bf16* row = s_ctx + r * ND;
        float mu = 0.f;
        for (int d = 0; d < ND; ++d) mu += (float)row[d];
        mu *= (1.f / (float)ND);
        float var = 0.f;
        for (int d = 0; d < ND; ++d) { float x = (float)row[d] - mu; var += x * x; }
        var *= (1.f / (float)ND);
        float rs = rsqrtf(var + 1e-5f);
        int b = b0 + r / NG, g = r % NG;
        __bf16* dst = h_out + (size_t)b * FLATD + g * ND;
        for (int d = 0; d < ND; ++d)
            dst[d] = (__bf16)(((float)row[d] - mu) * rs * s_ln[d] + s_ln[ND + d]);
    }
}

// ---------------------------------------------------------------------------
// Kernel M: fused 3-layer MLP, all GEMMs via WMMA bf16 with packed-B weights.
// 1024 threads = 32 waves; wave w -> row-tile (w>>2), col-quarter (w&3).
// ---------------------------------------------------------------------------
__global__ __launch_bounds__(1024) void mlp_kernel(
    const __bf16* __restrict__ h_g,
    const __bf16* __restrict__ w1p, const __bf16* __restrict__ w2p,
    const __bf16* __restrict__ w3p,
    const float* __restrict__ b1, const float* __restrict__ b2,
    const float* __restrict__ b3, float* __restrict__ out)
{
    __shared__ __bf16 sA[128 * 64];                                   //  16 KB
    __shared__ __attribute__((aligned(32))) __bf16 sB[64 * 512];      //  64 KB (reused as y2)
    __shared__ __bf16 sY1[128 * 512];                                 // 128 KB

    const int tid  = threadIdx.x;
    const int lane = tid & 31;
    const int w    = tid >> 5;
    const int rt   = w >> 2;             // 0..7 row tile
    const int cq   = w & 3;              // 0..3 col quarter
    const int n    = lane & 15;
    const int hi   = lane >> 4;
    const int m0   = blockIdx.x * 128;
    const f32x8 zero8 = {0.f, 0.f, 0.f, 0.f, 0.f, 0.f, 0.f, 0.f};

    // ---------------- layer 1: K = 3840 in 60 slabs of 64 ----------------
    f32x8 acc[8];
#pragma unroll
    for (int i = 0; i < 8; ++i) acc[i] = zero8;

    for (int kt = 0; kt < 60; ++kt) {
        unsigned* dA = (unsigned*)sA;
        for (int i = tid; i < 4096; i += 1024) {
            int r = i >> 5, kw = i & 31;
            dA[i] = ((const unsigned*)(h_g + (size_t)(m0 + r) * FLATD + kt * 64))[kw];
        }
        unsigned* dB = (unsigned*)sB;
        const unsigned* srcB = (const unsigned*)(w1p + (size_t)kt * 64 * N1);
        for (int i = tid; i < 16384; i += 1024) dB[i] = srcB[i];
        // prefetch next w1 slab (128 B cacheline per thread, 512 lines)
        if (kt + 1 < 60 && tid < 512)
            __builtin_prefetch(w1p + (size_t)(kt + 1) * 64 * N1 + (size_t)tid * 64, 0, 1);
        __syncthreads();

#pragma unroll
        for (int kk = 0; kk < 2; ++kk) {
            bf16x16 a = load_a_frag(sA + rt * 16 * 64 + kk * 32, 64, lane);
#pragma unroll
            for (int i = 0; i < 8; ++i) {
                int ct = cq * 8 + i;
                bf16x16 b = load_b_packed(sB, kk * 32 + ct, lane);
                acc[i] = wmma_bf16(a, b, acc[i]);
            }
        }
        __syncthreads();
    }
#pragma unroll
    for (int i = 0; i < 8; ++i) {
        int ct = cq * 8 + i;
#pragma unroll
        for (int j = 0; j < 8; ++j) {
            int r = rt * 16 + j + hi * 8;
            int c = ct * 16 + n;
            sY1[r * N1 + c] = (__bf16)fmaxf(acc[i][j] + b1[c], 0.f);
        }
    }
    __syncthreads();

    // ---------------- layer 2: [128x512] @ [512x256] ----------------
    f32x8 acc2[4];
#pragma unroll
    for (int i = 0; i < 4; ++i) acc2[i] = zero8;
    for (int kk = 0; kk < 16; ++kk) {
        bf16x16 a = load_a_frag(sY1 + rt * 16 * N1 + kk * 32, N1, lane);
#pragma unroll
        for (int i = 0; i < 4; ++i) {
            int ct = cq * 4 + i;
            bf16x16 b = load_b_packed(w2p, kk * 16 + ct, lane);
            acc2[i] = wmma_bf16(a, b, acc2[i]);
        }
    }
    __syncthreads();                     // before reusing sB as y2
    __bf16* sY2 = sB;
#pragma unroll
    for (int i = 0; i < 4; ++i) {
        int ct = cq * 4 + i;
#pragma unroll
        for (int j = 0; j < 8; ++j) {
            int r = rt * 16 + j + hi * 8;
            int c = ct * 16 + n;
            sY2[r * N2 + c] = (__bf16)fmaxf(acc2[i][j] + b2[c], 0.f);
        }
    }
    __syncthreads();

    // ---------------- layer 3: [128x256] @ [256x256] ----------------
    f32x8 acc3[4];
#pragma unroll
    for (int i = 0; i < 4; ++i) acc3[i] = zero8;
    for (int kk = 0; kk < 8; ++kk) {
        bf16x16 a = load_a_frag(sY2 + rt * 16 * N2 + kk * 32, N2, lane);
#pragma unroll
        for (int i = 0; i < 4; ++i) {
            int ct = cq * 4 + i;
            bf16x16 b = load_b_packed(w3p, kk * 16 + ct, lane);
            acc3[i] = wmma_bf16(a, b, acc3[i]);
        }
    }
#pragma unroll
    for (int i = 0; i < 4; ++i) {
        int ct = cq * 4 + i;
#pragma unroll
        for (int j = 0; j < 8; ++j) {
            int r = rt * 16 + j + hi * 8;
            int c = ct * 16 + n;
            out[(size_t)(m0 + r) * N3 + c] = acc3[i][j] + b3[c];
        }
    }
}

// ---------------------------------------------------------------------------
extern "C" void kernel_launch(void* const* d_in, const int* in_sizes, int n_in,
                              void* d_out, int out_size, void* d_ws, size_t ws_size,
                              hipStream_t stream) {
    (void)in_sizes; (void)n_in; (void)out_size; (void)ws_size;
    const float* gf    = (const float*)d_in[0];
    const float* gene  = (const float*)d_in[1];
    const float* typee = (const float*)d_in[2];
    const float* wbin  = (const float*)d_in[3];
    const float* bbin  = (const float*)d_in[4];
    const float* wfeat = (const float*)d_in[5];
    const float* bfeat = (const float*)d_in[6];
    const float* ipw   = (const float*)d_in[7];
    const float* ipb   = (const float*)d_in[8];
    const float* outw  = (const float*)d_in[9];
    const float* outb  = (const float*)d_in[10];
    const float* lng   = (const float*)d_in[11];
    const float* lnb   = (const float*)d_in[12];
    const float* w1    = (const float*)d_in[13];
    const float* b1    = (const float*)d_in[14];
    const float* w2    = (const float*)d_in[15];
    const float* b2    = (const float*)d_in[16];
    const float* w3    = (const float*)d_in[17];
    const float* b3    = (const float*)d_in[18];

    char* ws = (char*)d_ws;
    __bf16* h_g  = (__bf16*)(ws + H_OFF);
    __bf16* w1p  = (__bf16*)(ws + W1T_OFF);
    __bf16* w2p  = (__bf16*)(ws + W2T_OFF);
    __bf16* w3p  = (__bf16*)(ws + W3T_OFF);
    __bf16* owp  = (__bf16*)(ws + OWT_OFF);
    __bf16* qkvb = (__bf16*)(ws + QKVB_OFF);
    float*  u    = (float*)(ws + U_OFF);
    float*  ag   = (float*)(ws + AG_OFF);
    float*  e    = (float*)(ws + E_OFF);
    float*  out  = (float*)d_out;

    prep_kernel<<<64, 256, 0, stream>>>(gene, typee, wbin, bbin, wfeat, bfeat,
                                        ipw, ipb, outw, w1, w2, w3,
                                        w1p, w2p, w3p, owp, qkvb, u, ag, e);
    attn_kernel<<<BATCH / 16, 256, 0, stream>>>(gf, outb, lng, lnb,
                                                qkvb, u, ag, e, owp, h_g);
    mlp_kernel<<<BATCH / 128, 1024, 0, stream>>>(h_g, w1p, w2p, w3p,
                                                 b1, b2, b3, out);
}
